// TemporalCausalTransformer_60318520705657
// MI455X (gfx1250) — matrix operand order repaired and checked by reading, not compiled, not run on hardware
//
#include <hip/hip_runtime.h>
#include <math.h>

#define B_   32
#define S_   512
#define V_   50
#define D_   256
#define H_   8
#define FF_  1024
#define HD_  32
#define LAG_ 10
#define NL_  4
#define NS_  4
#define BS_  (B_ * S_)

typedef __attribute__((ext_vector_type(16))) _Float16  v16h;
typedef __attribute__((ext_vector_type(8)))  float     v8f;
typedef __attribute__((ext_vector_type(8)))  unsigned  v8u;

// pack two f32 -> packed f16 pair (low = x, high = y); v_cvt_pkrtz_f16_f32
static __device__ __forceinline__ unsigned pack_f16(float x, float y) {
  auto p = __builtin_amdgcn_cvt_pkrtz(x, y);   // __fp16 ext_vector(2)
  return __builtin_bit_cast(unsigned, p);
}

static __device__ __forceinline__ float rcp_fast(float x) {
  return __builtin_amdgcn_rcpf(x);
}
static __device__ __forceinline__ float sigmoid_fast(float x) {
  return rcp_fast(1.0f + __expf(-x));
}
static __device__ __forceinline__ float tanh_fast(float x) {
  float e = __expf(2.0f * x);
  return (e - 1.0f) * rcp_fast(e + 1.0f);
}
template <int ACT>
static __device__ __forceinline__ float act_apply(float v) {
  if      (ACT == 1) v = fmaxf(v, 0.0f);
  else if (ACT == 2) v = tanh_fast(v);
  else if (ACT == 3) v = sigmoid_fast(v);
  return v;
}

// ---------------------------------------------------------------------------
// FAST WMMA GEMM: requires M%32==0, N%32==0, K%32==0. No bounds checks.
// One wave computes a 32x32 C tile (2x2 sub-tiles of 16x16, 4 WMMA/K-step).
// C[m,n] = act( sum_k A[m,k]*Bw[k,n] + bias[n] ); f16 multiply, f32 acc.
// ---------------------------------------------------------------------------
template <int ACT>
__global__ void gemm_fast(const float* __restrict__ A, int lda,
                          const float* __restrict__ Bw, int ldb,
                          const float* __restrict__ bias,
                          float* __restrict__ C, int ldc,
                          int K)
{
  const int lane = threadIdx.x;
  const int half = lane >> 4;
  const int l16  = lane & 15;
  const int m0 = blockIdx.y * 32;
  const int n0 = blockIdx.x * 32;

  const float* Ar0 = A + (size_t)(m0 + l16) * lda;
  const float* Ar1 = Ar0 + (size_t)16 * lda;
  const int nb0 = n0 + l16;
  const int nb1 = nb0 + 16;

  v8f acc00 = {}, acc01 = {}, acc10 = {}, acc11 = {};

  for (int kk = 0; kk < K; kk += 32) {
    v8u a0u, a1u, b0u, b1u;
#pragma unroll
    for (int v = 0; v < 8; ++v) {
      const int ka = kk + half * 16 + 2 * v;
      float2 p0 = *(const float2*)(Ar0 + ka);
      float2 p1 = *(const float2*)(Ar1 + ka);
      a0u[v] = pack_f16(p0.x, p0.y);
      a1u[v] = pack_f16(p1.x, p1.y);
      const int kb = kk + half * 16 + 2 * v;
      const float* Br0 = Bw + (size_t)kb * ldb;
      const float* Br1 = Br0 + ldb;
      b0u[v] = pack_f16(Br0[nb0], Br1[nb0]);
      b1u[v] = pack_f16(Br0[nb1], Br1[nb1]);
    }
    // prefetch next K-step (global_prefetch_b8; speculative, safe past end)
    __builtin_prefetch((const void*)(Ar0 + kk + 32), 0, 1);
    __builtin_prefetch((const void*)(Ar1 + kk + 32), 0, 1);
    __builtin_prefetch((const void*)(Bw + (size_t)(kk + 32 + half * 16) * ldb + nb0), 0, 1);

    v16h a0 = __builtin_bit_cast(v16h, a0u);
    v16h a1 = __builtin_bit_cast(v16h, a1u);
    v16h b0 = __builtin_bit_cast(v16h, b0u);
    v16h b1 = __builtin_bit_cast(v16h, b1u);
    acc00 = __builtin_amdgcn_wmma_f32_16x16x32_f16(false, a0, false, b0, (short)0, acc00, false, false);
    acc01 = __builtin_amdgcn_wmma_f32_16x16x32_f16(false, a0, false, b1, (short)0, acc01, false, false);
    acc10 = __builtin_amdgcn_wmma_f32_16x16x32_f16(false, a1, false, b0, (short)0, acc10, false, false);
    acc11 = __builtin_amdgcn_wmma_f32_16x16x32_f16(false, a1, false, b1, (short)0, acc11, false, false);
  }

  const float bv0 = bias[nb0];
  const float bv1 = bias[nb1];
#pragma unroll
  for (int r = 0; r < 8; ++r) {
    const int mr0 = m0 + r + half * 8;
    const int mr1 = mr0 + 16;
    C[(size_t)mr0 * ldc + nb0] = act_apply<ACT>(acc00[r] + bv0);
    C[(size_t)mr0 * ldc + nb1] = act_apply<ACT>(acc01[r] + bv1);
    C[(size_t)mr1 * ldc + nb0] = act_apply<ACT>(acc10[r] + bv0);
    C[(size_t)mr1 * ldc + nb1] = act_apply<ACT>(acc11[r] + bv1);
  }
}

// ---------------------------------------------------------------------------
// BOUNDED WMMA GEMM: M%16==0, K%32==0, N ragged. 16x16 tile per wave.
// N-guards use clamped addresses + selects (no exec-mask branches).
// ---------------------------------------------------------------------------
template <int ACT>
__global__ void gemm_bounded(const float* __restrict__ A, int lda,
                             const float* __restrict__ Bw, int ldb,
                             const float* __restrict__ bias,
                             float* __restrict__ C, int ldc,
                             int N, int K)
{
  const int lane = threadIdx.x;
  const int half = lane >> 4;
  const int l16  = lane & 15;
  const int m0 = blockIdx.y * 16;
  const int n0 = blockIdx.x * 16;

  const float* Arow = A + (size_t)(m0 + l16) * lda;
  const int nb = n0 + l16;
  const bool bn = (nb < N);
  const int nbc = bn ? nb : (N - 1);      // clamped column, select later

  v8f acc = {};
  for (int kk = 0; kk < K; kk += 32) {
    v8u au, bu;
#pragma unroll
    for (int v = 0; v < 8; ++v) {
      const int ka = kk + half * 16 + 2 * v;
      float2 pa = *(const float2*)(Arow + ka);
      au[v] = pack_f16(pa.x, pa.y);
      const int kb = kk + half * 16 + 2 * v;
      float b0 = Bw[(size_t)kb * ldb + nbc];
      float b1 = Bw[(size_t)(kb + 1) * ldb + nbc];
      b0 = bn ? b0 : 0.0f;
      b1 = bn ? b1 : 0.0f;
      bu[v] = pack_f16(b0, b1);
    }
    v16h af = __builtin_bit_cast(v16h, au);
    v16h bf = __builtin_bit_cast(v16h, bu);
    acc = __builtin_amdgcn_wmma_f32_16x16x32_f16(false, af, false, bf,
                                                 (short)0, acc, false, false);
  }

  const float bv = bias[nbc];
#pragma unroll
  for (int r = 0; r < 8; ++r) {
    const int m = m0 + r + half * 8;
    float v = act_apply<ACT>(acc[r] + bv);
    if (bn) C[(size_t)m * ldc + nb] = v;
  }
}

// ---------------------------------------------------------------------------
// Input projection + sinusoidal positional encoding.
// ---------------------------------------------------------------------------
__global__ void input_proj_pe(const float* __restrict__ x,
                              const float* __restrict__ W,
                              const float* __restrict__ bias,
                              float* __restrict__ h)
{
  __shared__ float xr[V_];
  const int row = blockIdx.x;          // b*S + s
  const int d   = threadIdx.x;         // 0..255
  const int s   = row % S_;
  if (d < V_) xr[d] = x[(size_t)row * V_ + d];
  __syncthreads();
  float acc = bias[d];
  for (int j = 0; j < V_; ++j) acc += xr[j] * W[(size_t)j * D_ + d];
  const int jj = d & ~1;
  float arg = (float)s * __expf(-(float)jj * 0.03597789152f); // ln(1e4)/256
  acc += (d & 1) ? __cosf(arg) : __sinf(arg);
  h[(size_t)row * D_ + d] = acc;
}

// conv1d D->D/2, k=3, pad=1, relu; c1 stored (B,S,128)
__global__ void conv1_relu(const float* __restrict__ h,
                           const float* __restrict__ w,
                           const float* __restrict__ bias,
                           float* __restrict__ c1)
{
  __shared__ float rows[3 * D_];
  const int bs = blockIdx.x;
  const int b = bs / S_, s = bs % S_;
  const int oc = threadIdx.x;          // 0..127
  for (int i = threadIdx.x; i < 3 * D_; i += blockDim.x) {
    int t = i / D_, ic = i % D_;
    int ss = s + t - 1;
    rows[i] = (ss >= 0 && ss < S_) ? h[((size_t)b * S_ + ss) * D_ + ic] : 0.0f;
  }
  __syncthreads();
  float acc = bias[oc];
  for (int ic = 0; ic < D_; ++ic) {
    const float* wp = w + ((size_t)oc * D_ + ic) * 3;
    acc += rows[ic] * wp[0] + rows[D_ + ic] * wp[1] + rows[2 * D_ + ic] * wp[2];
  }
  c1[((size_t)b * S_ + s) * 128 + oc] = fmaxf(acc, 0.0f);
}

// conv1d 128->1, k=3, pad=1, sigmoid; writes change ws buffer AND d_out slice
__global__ void conv2_sig(const float* __restrict__ c1,
                          const float* __restrict__ w,
                          const float* __restrict__ bias,
                          float* __restrict__ change,
                          float* __restrict__ out_change)
{
  const int bs = blockIdx.x * blockDim.x + threadIdx.x;
  if (bs >= B_ * S_) return;
  const int b = bs / S_, s = bs % S_;
  float acc = bias[0];
  for (int t = 0; t < 3; ++t) {
    int ss = s + t - 1;
    if (ss < 0 || ss >= S_) continue;
    const float* cp = c1 + ((size_t)b * S_ + ss) * 128;
    for (int ic = 0; ic < 128; ++ic) acc += cp[ic] * w[ic * 3 + t];
  }
  float v = sigmoid_fast(acc);
  change[bs]     = v;
  out_change[bs] = v;
}

// sequential adaptive smoothing scan over S (in-place on h)
__global__ void scan_kernel(float* __restrict__ h,
                            const float* __restrict__ adapt,
                            const float* __restrict__ change,
                            const float* __restrict__ smooth)
{
  const int b = blockIdx.x;
  const int d = threadIdx.x;
  const float sw = sigmoid_fast(smooth[0]);
  float a = h[((size_t)b * S_) * D_ + d];
  for (int t = 1; t < S_; ++t) {
    size_t idx = ((size_t)b * S_ + t) * D_ + d;
    float c  = change[b * S_ + t];
    float xt = h[idx];
    a = c * adapt[idx] + (1.0f - c) * (sw * a + (1.0f - sw) * xt);
    h[idx] = a;
  }
}

// h = LayerNorm(h + res) * g + b   (one 256-thread block per row)
__global__ void add_ln(float* __restrict__ h, const float* __restrict__ res,
                       const float* __restrict__ g, const float* __restrict__ b)
{
  __shared__ float red[8];
  const int row = blockIdx.x;
  const int tid = threadIdx.x;
  const size_t idx = (size_t)row * D_ + tid;
  float x = h[idx] + res[idx];

  float s = x;
  for (int m = 1; m < 32; m <<= 1) s += __shfl_xor(s, m, 32);
  if ((tid & 31) == 0) red[tid >> 5] = s;
  __syncthreads();
  float tot = 0.0f;
#pragma unroll
  for (int i = 0; i < 8; ++i) tot += red[i];
  float mean = tot * (1.0f / D_);
  __syncthreads();

  float d = x - mean;
  float s2 = d * d;
  for (int m = 1; m < 32; m <<= 1) s2 += __shfl_xor(s2, m, 32);
  if ((tid & 31) == 0) red[tid >> 5] = s2;
  __syncthreads();
  float tv = 0.0f;
#pragma unroll
  for (int i = 0; i < 8; ++i) tv += red[i];
  float var = tv * (1.0f / D_);
  h[idx] = d * rsqrtf(var + 1e-5f) * g[tid] + b[tid];
}

// ---------------------------------------------------------------------------
// Fused banded-causal attention (flash-style), one wave per (b, h, 16-q-rows).
// Decay mask kills lag>LAG, so only 1-2 K-tiles of 32 are ever visited.
// ---------------------------------------------------------------------------
__global__ void attn_fused(const float* __restrict__ Q,
                           const float* __restrict__ Kt,
                           const float* __restrict__ Vt,
                           const float* __restrict__ decay_p,
                           float* __restrict__ O)
{
  __shared__ float Pt[16 * 32];
  __shared__ float dsh[16];
  const int lane = threadIdx.x;
  const int half = lane >> 4;
  const int l16  = lane & 15;
  const int qt = blockIdx.x, hh = blockIdx.y, bb = blockIdx.z;
  const int q0 = qt * 16;

  if (lane <= LAG_) {
    float sp = log1pf(__expf(decay_p[lane]));   // softplus
    dsh[lane] = -logf(sp + 1e-8f);
  }
  __syncthreads();

  // Q A-fragment (constant: K-dim == HD == 32, exactly one WMMA step)
  v16h qa;
  {
    const float* qp = Q + ((size_t)bb * S_ + (q0 + l16)) * D_ + hh * HD_;
    v8u qu;
#pragma unroll
    for (int v = 0; v < 8; ++v) {
      int d = half * 16 + 2 * v;
      float2 p = *(const float2*)(qp + d);
      qu[v] = pack_f16(p.x, p.y);
    }
    qa = __builtin_bit_cast(v16h, qu);
  }

  float mi[8], li[8];
#pragma unroll
  for (int r = 0; r < 8; ++r) { mi[r] = -1e30f; li[r] = 0.0f; }
  v8f o0 = {}, o1 = {};
  const float inv_scale = 0.17677669529f;   // 1/sqrt(32)

  int kt0 = (q0 > LAG_) ? (q0 - LAG_) / 32 : 0;
  int kt1 = (q0 + 15) / 32;
  for (int kt = kt0; kt <= kt1; ++kt) {
    const int kbase = kt * 32;

    // scores for 32 K positions: two 16-wide score tiles
    v8f sc[2];
#pragma unroll
    for (int sub = 0; sub < 2; ++sub) {
      const float* kpp = Kt + ((size_t)bb * S_ + (kbase + sub * 16 + l16)) * D_ + hh * HD_;
      v8u ku;
#pragma unroll
      for (int v = 0; v < 8; ++v) {
        int d = half * 16 + 2 * v;
        float2 p = *(const float2*)(kpp + d);
        ku[v] = pack_f16(p.x, p.y);
      }
      v16h kb = __builtin_bit_cast(v16h, ku);
      v8f z = {};
      sc[sub] = __builtin_amdgcn_wmma_f32_16x16x32_f16(false, qa, false, kb,
                                                       (short)0, z, false, false);
    }

    // mask + online softmax over the 32 columns, stage P into LDS
#pragma unroll
    for (int r = 0; r < 8; ++r) {
      int qrow = q0 + r + half * 8;
      int kc0 = kbase + l16;
      int kc1 = kbase + 16 + l16;
      int lag0 = qrow - kc0, lag1 = qrow - kc1;
      float s0 = sc[0][r] * inv_scale +
                 ((lag0 < 0 || lag0 > LAG_) ? -1e9f : dsh[lag0]);
      float s1 = sc[1][r] * inv_scale +
                 ((lag1 < 0 || lag1 > LAG_) ? -1e9f : dsh[lag1]);
      float rmax = fmaxf(s0, s1);
      rmax = fmaxf(rmax, __shfl_xor(rmax, 1, 32));
      rmax = fmaxf(rmax, __shfl_xor(rmax, 2, 32));
      rmax = fmaxf(rmax, __shfl_xor(rmax, 4, 32));
      rmax = fmaxf(rmax, __shfl_xor(rmax, 8, 32));
      float mnew = fmaxf(mi[r], rmax);
      float p0 = __expf(s0 - mnew);
      float p1 = __expf(s1 - mnew);
      float ps = p0 + p1;
      ps += __shfl_xor(ps, 1, 32);
      ps += __shfl_xor(ps, 2, 32);
      ps += __shfl_xor(ps, 4, 32);
      ps += __shfl_xor(ps, 8, 32);
      float corr = __expf(mi[r] - mnew);
      li[r] = li[r] * corr + ps;
      mi[r] = mnew;
      o0[r] *= corr;
      o1[r] *= corr;
      Pt[(r + half * 8) * 32 + l16]      = p0;
      Pt[(r + half * 8) * 32 + 16 + l16] = p1;
    }
    __syncthreads();

    // P (16x32) as A-fragment
    v16h pa;
    {
      v8u pu;
#pragma unroll
      for (int v = 0; v < 8; ++v) {
        int kk = half * 16 + 2 * v;
        pu[v] = pack_f16(Pt[l16 * 32 + kk], Pt[l16 * 32 + kk + 1]);
      }
      pa = __builtin_bit_cast(v16h, pu);
    }
    __syncthreads();

    // context accumulation: V as B-fragment, two HD halves
#pragma unroll
    for (int nh = 0; nh < 2; ++nh) {
      int d = nh * 16 + l16;
      v8u vu;
#pragma unroll
      for (int v = 0; v < 8; ++v) {
        int kp = kbase + half * 16 + 2 * v;
        float v0 = Vt[((size_t)bb * S_ + kp) * D_ + hh * HD_ + d];
        float v1 = Vt[((size_t)bb * S_ + kp + 1) * D_ + hh * HD_ + d];
        vu[v] = pack_f16(v0, v1);
      }
      v16h vb = __builtin_bit_cast(v16h, vu);
      if (nh == 0)
        o0 = __builtin_amdgcn_wmma_f32_16x16x32_f16(false, pa, false, vb,
                                                    (short)0, o0, false, false);
      else
        o1 = __builtin_amdgcn_wmma_f32_16x16x32_f16(false, pa, false, vb,
                                                    (short)0, o1, false, false);
    }
  }

#pragma unroll
  for (int r = 0; r < 8; ++r) {
    int qrow = q0 + r + half * 8;
    float inv = rcp_fast(li[r]);
    float* op = O + ((size_t)bb * S_ + qrow) * D_ + hh * HD_;
    op[l16]      = o0[r] * inv;
    op[16 + l16] = o1[r] * inv;
  }
}

// ---------------------------------------------------------------------------
extern "C" void kernel_launch(void* const* d_in, const int* in_sizes, int n_in,
                              void* d_out, int out_size, void* d_ws, size_t ws_size,
                              hipStream_t stream) {
  (void)in_sizes; (void)n_in; (void)out_size; (void)ws_size;
  const float* x      = (const float*)d_in[0];
  const float* W_in   = (const float*)d_in[1];
  const float* b_in   = (const float*)d_in[2];
  const float* cd_w1  = (const float*)d_in[3];
  const float* cd_b1  = (const float*)d_in[4];
  const float* cd_w2  = (const float*)d_in[5];
  const float* cd_b2  = (const float*)d_in[6];
  const float* ad_w1  = (const float*)d_in[7];
  const float* ad_b1  = (const float*)d_in[8];
  const float* ad_w2  = (const float*)d_in[9];
  const float* ad_b2  = (const float*)d_in[10];
  const float* smooth = (const float*)d_in[11];
  const float* sc_w1  = (const float*)d_in[12];
  const float* sc_b1  = (const float*)d_in[13];
  const float* sc_w2  = (const float*)d_in[14];
  const float* sc_b2  = (const float*)d_in[15];
  const float* fus_w1 = (const float*)d_in[16];
  const float* fus_b1 = (const float*)d_in[17];
  const float* fus_w2 = (const float*)d_in[18];
  const float* fus_b2 = (const float*)d_in[19];
  const float* inst_w1= (const float*)d_in[20];
  const float* inst_b1= (const float*)d_in[21];
  const float* inst_w2= (const float*)d_in[22];
  const float* inst_b2= (const float*)d_in[23];
  const float* int_w1 = (const float*)d_in[24];
  const float* int_b1 = (const float*)d_in[25];
  const float* int_w2 = (const float*)d_in[26];
  const float* int_b2 = (const float*)d_in[27];
  auto LW = [&](int l, int j) { return (const float*)d_in[28 + l * 17 + j]; };

  float* out = (float*)d_out;
  float* ws  = (float*)d_ws;
  // workspace layout (floats)
  float* h      = ws;                                   // BS*D
  float* qbuf   = h      + (size_t)BS_ * D_;            // BS*D (adapt, then q)
  float* kbuf   = qbuf   + (size_t)BS_ * D_;            // BS*D
  float* vbuf   = kbuf   + (size_t)BS_ * D_;            // BS*D
  float* ebuf   = vbuf   + (size_t)BS_ * D_;            // BS*D (attn_out / ff_out)
  float* fbuf   = ebuf   + (size_t)BS_ * D_;            // BS*FF (tanh-mid / ctx / ff-mid)
  float* c1     = fbuf   + (size_t)BS_ * FF_;           // B*S*128
  float* change = c1     + (size_t)BS_ * 128;           // B*S
  float* cat    = change + (size_t)BS_;                 // B*100000
  float* p2     = cat    + (size_t)B_ * 100000;         // B*D
  float* p3     = p2     + (size_t)B_ * D_;             // B*128
  float* s1     = p3     + (size_t)B_ * 128;            // B*128

  // 1) input projection + positional encoding
  input_proj_pe<<<BS_, D_, 0, stream>>>(x, W_in, b_in, h);

  // 2) adaptation MLP: adapt = tanh(h@ad_w1+b)@ad_w2+b
  dim3 gDD(D_ / 32, BS_ / 32);
  gemm_fast<2><<<gDD, 32, 0, stream>>>(h,    D_, ad_w1, D_, ad_b1, fbuf, D_, D_);
  gemm_fast<0><<<gDD, 32, 0, stream>>>(fbuf, D_, ad_w2, D_, ad_b2, qbuf, D_, D_);

  // 3) change detector convs (writes change + final output slice)
  conv1_relu<<<BS_, 128, 0, stream>>>(h, cd_w1, cd_b1, c1);
  conv2_sig<<<(BS_ + 255) / 256, 256, 0, stream>>>(c1, cd_w2, cd_b2, change, out + 881600);

  // 4) sequential adaptive scan (in-place on h); adapt currently in qbuf
  scan_kernel<<<B_, D_, 0, stream>>>(h, qbuf, change, smooth);

  // 5) transformer layers
  for (int l = 0; l < NL_; ++l) {
    gemm_fast<0><<<gDD, 32, 0, stream>>>(h, D_, LW(l, 0), D_, LW(l, 1), qbuf, D_, D_);
    gemm_fast<0><<<gDD, 32, 0, stream>>>(h, D_, LW(l, 2), D_, LW(l, 3), kbuf, D_, D_);
    gemm_fast<0><<<gDD, 32, 0, stream>>>(h, D_, LW(l, 4), D_, LW(l, 5), vbuf, D_, D_);
    attn_fused<<<dim3(S_ / 16, H_, B_), 32, 0, stream>>>(qbuf, kbuf, vbuf, LW(l, 8), fbuf);
    gemm_fast<0><<<gDD, 32, 0, stream>>>(fbuf, D_, LW(l, 6), D_, LW(l, 7), ebuf, D_, D_);
    add_ln<<<BS_, D_, 0, stream>>>(h, ebuf, LW(l, 13), LW(l, 14));
    dim3 gDF(FF_ / 32, BS_ / 32);
    gemm_fast<1><<<gDF, 32, 0, stream>>>(h,    D_,  LW(l, 9),  FF_, LW(l, 10), fbuf, FF_, D_);
    gemm_fast<0><<<gDD, 32, 0, stream>>>(fbuf, FF_, LW(l, 11), D_,  LW(l, 12), ebuf, D_,  FF_);
    add_ln<<<BS_, D_, 0, stream>>>(h, ebuf, LW(l, 15), LW(l, 16));
  }

  // 6) heads; pooled = h[:, S-1, :] via strided view (lda = S*D)
  const float* pooled = h + (size_t)(S_ - 1) * D_;
  const int ldp = S_ * D_;
  dim3 g128(128 / 32, B_ / 32);                 // fast: M=32, N=128
  dim3 g25k((25000 + 15) / 16, B_ / 16);        // bounded
  for (int i = 0; i < NS_; ++i) {
    gemm_fast<1><<<g128, 32, 0, stream>>>(pooled, ldp,
        sc_w1 + (size_t)i * D_ * 128, 128, sc_b1 + (size_t)i * 128,
        s1, 128, D_);
    gemm_bounded<3><<<g25k, 32, 0, stream>>>(s1, 128,
        sc_w2 + (size_t)i * 128 * 25000, 25000, sc_b2 + (size_t)i * 25000,
        cat + (size_t)i * 25000, 100000, 25000, 128);
  }
  // fusion: temporal -> out[0 .. 800000)
  dim3 gD(D_ / 32, B_ / 32);
  gemm_fast<1><<<gD, 32, 0, stream>>>(cat, 100000, fus_w1, D_, fus_b1, p2, D_, 100000);
  gemm_bounded<3><<<g25k, 32, 0, stream>>>(p2, D_, fus_w2, 25000, fus_b2, out, 25000, 25000, D_);
  // inst -> out[800000 .. 880000)
  gemm_fast<1><<<g128, 32, 0, stream>>>(pooled, ldp, inst_w1, 128, inst_b1, p3, 128, D_);
  dim3 g2500((2500 + 15) / 16, B_ / 16);
  gemm_bounded<3><<<g2500, 32, 0, stream>>>(p3, 128, inst_w2, 2500, inst_b2, out + 800000, 2500, 2500, 128);
  // interv -> out[880000 .. 881600)
  gemm_fast<1><<<g128, 32, 0, stream>>>(pooled, ldp, int_w1, 128, int_b1, p3, 128, D_);
  dim3 g50((50 + 15) / 16, B_ / 16);
  gemm_bounded<3><<<g50, 32, 0, stream>>>(p3, 128, int_w2, 50, int_b2, out + 880000, 50, 50, 128);
  // change scores already written at out+881600 by conv2_sig
}
